// MultiHeadAttentionWithRoPE_78735340470829
// MI455X (gfx1250) — matrix operand order repaired and checked by reading, not compiled, not run on hardware
//
#include <hip/hip_runtime.h>
#include <hip/hip_bf16.h>
#include <math.h>

typedef __attribute__((ext_vector_type(16))) __bf16 v16bf;
typedef __attribute__((ext_vector_type(8)))  __bf16 v8bf;
typedef __attribute__((ext_vector_type(8)))  float  v8f;
typedef __attribute__((ext_vector_type(4)))  int    v4i;

#define WMMA_BF16(a, b, c) \
  __builtin_amdgcn_wmma_f32_16x16x32_bf16(false, (a), false, (b), (short)0, (c), false, false)

// CDNA5 async global->LDS path (ASYNCcnt), guarded so absence of the builtin
// falls back to plain LDS stores instead of failing the build.
#if defined(__has_builtin)
#  if __has_builtin(__builtin_amdgcn_global_load_async_to_lds_b128) && \
      __has_builtin(__builtin_amdgcn_s_wait_asynccnt)
#    define HAVE_ASYNC_LDS 1
#  endif
#endif
#ifndef HAVE_ASYNC_LDS
#  define HAVE_ASYNC_LDS 0
#endif

#if HAVE_ASYNC_LDS
typedef __attribute__((address_space(1))) v4i* as1_v4i_ptr;
typedef __attribute__((address_space(3))) v4i* as3_v4i_ptr;
__device__ __forceinline__ void async_b128_to_lds(const __bf16* gsrc, __bf16* ldst) {
  __builtin_amdgcn_global_load_async_to_lds_b128(
      (as1_v4i_ptr)(gsrc), (as3_v4i_ptr)(ldst), 0, 0);
}
#endif

// Problem shape (fixed by setup_inputs)
static constexpr int Bn = 2, Ss = 2048, Hh = 16, HD = 64, Dm = 1024;
static constexpr int Mrows = Bn * Ss;        // 4096
static constexpr int N3 = 3 * Dm;            // 3072

// ---------------------------------------------------------------------------
// Elementwise converters
// ---------------------------------------------------------------------------
__global__ void k_cvt(const float* __restrict__ x, __bf16* __restrict__ y, int n) {
  int i = blockIdx.x * blockDim.x + threadIdx.x;
  if (i < n) y[i] = (__bf16)x[i];
}

// WT[n*K + k] = bf16(W[k*N + n])   (weight transpose+convert: B-fragment friendly)
__global__ void k_wtrans(const float* __restrict__ W, __bf16* __restrict__ WT, int K, int N) {
  int i = blockIdx.x * blockDim.x + threadIdx.x;
  if (i >= K * N) return;
  int n = i / K, k = i % K;
  WT[i] = (__bf16)W[(size_t)k * N + n];
}

__global__ void k_resid(const float* __restrict__ a, const float* __restrict__ b,
                        float* __restrict__ yf, __bf16* __restrict__ yb, int n) {
  int i = blockIdx.x * blockDim.x + threadIdx.x;
  if (i < n) { float v = a[i] + b[i]; yf[i] = v; yb[i] = (__bf16)v; }
}

__global__ void k_silu(const float* __restrict__ x, __bf16* __restrict__ y, int n) {
  int i = blockIdx.x * blockDim.x + threadIdx.x;
  if (i < n) { float v = x[i]; y[i] = (__bf16)(v / (1.f + expf(-v))); }
}

// ---------------------------------------------------------------------------
// Row-wise l2norm-scale: y = w * v / (||v|| + eps), v = X (+ X2 if non-null)
// ---------------------------------------------------------------------------
__global__ void __launch_bounds__(256) k_l2norm(const float* __restrict__ X,
                                                const float* __restrict__ X2,
                                                const float* __restrict__ w,
                                                float* __restrict__ Yf,
                                                __bf16* __restrict__ Yb, int C) {
  const int row = blockIdx.x;
  const size_t base = (size_t)row * C;
  float ss = 0.f;
  for (int c = threadIdx.x; c < C; c += blockDim.x) {
    float v = X[base + c];
    if (X2) v += X2[base + c];
    ss += v * v;
  }
  #pragma unroll
  for (int off = 16; off >= 1; off >>= 1) ss += __shfl_xor(ss, off, 32);
  __shared__ float red[8];
  if ((threadIdx.x & 31) == 0) red[threadIdx.x >> 5] = ss;
  __syncthreads();
  float tot = 0.f;
  #pragma unroll
  for (int i = 0; i < 8; ++i) tot += red[i];
  const float inv = 1.f / (sqrtf(tot) + 1e-8f);
  for (int c = threadIdx.x; c < C; c += blockDim.x) {
    float v = X[base + c];
    if (X2) v += X2[base + c];
    v = w[c] * v * inv;
    if (Yf) Yf[base + c] = v;
    if (Yb) Yb[base + c] = (__bf16)v;
  }
}

// ---------------------------------------------------------------------------
// GEMM: C[M,N] = A_bf16[M,K] @ W (given as WT_bf16[N,K]) + bias, f32 out.
// Block = 256 threads = 8 waves; block tile 256 rows x 64 cols.
// Wave wv -> rows [by*256 + 32*wv, +32) (2 A-fragments, 8 accumulators).
// The 64x32 B tile is shared by all 8 waves: staged into LDS (double-buffered)
// via CDNA5 async global->LDS loads when available.
// ---------------------------------------------------------------------------
__global__ void __launch_bounds__(256) k_gemm(const __bf16* __restrict__ A,
                                              const __bf16* __restrict__ WT,
                                              const float* __restrict__ bias,
                                              float* __restrict__ C,
                                              int M, int N, int K) {
  __shared__ __bf16 bt[2][64][32];  // [buf][n within col tile][k within k-step]
  const int tid = threadIdx.x;
  const int lane = tid & 31, wv = tid >> 5;
  const int lo = lane & 15, hi = lane >> 4;
  const int rowbase = blockIdx.y * 256 + wv * 32;
  const int colbase = blockIdx.x * 64;
  const __bf16* arow0 = A + (size_t)(rowbase + lo) * K;
  const __bf16* arow1 = A + (size_t)(rowbase + 16 + lo) * K;

  // staging assignment: each of 256 threads moves one 16B chunk of the B tile
  const int sn = tid >> 2;            // 0..63  (col within tile)
  const int sch = (tid & 3) * 8;      // 0,8,16,24 (k offset, 8 bf16 = 16B)
  const __bf16* sbase = WT + (size_t)(colbase + sn) * K + sch;

  v8f acc[2][4];
  #pragma unroll
  for (int rg = 0; rg < 2; ++rg)
    #pragma unroll
    for (int g = 0; g < 4; ++g)
      #pragma unroll
      for (int r = 0; r < 8; ++r) acc[rg][g][r] = 0.f;

  // prologue: stage first B tile
#if HAVE_ASYNC_LDS
  async_b128_to_lds(sbase, &bt[0][sn][sch]);
  __builtin_amdgcn_s_wait_asynccnt(0);
#else
  *(v8bf*)&bt[0][sn][sch] = *(const v8bf*)(sbase);
#endif
  __syncthreads();

  int buf = 0;
  for (int k0 = 0; k0 < K; k0 += 32, buf ^= 1) {
    // stage next k-step's B tile into the other buffer (overlap with compute)
    if (k0 + 32 < K) {
#if HAVE_ASYNC_LDS
      async_b128_to_lds(sbase + k0 + 32, &bt[buf ^ 1][sn][sch]);
#else
      *(v8bf*)&bt[buf ^ 1][sn][sch] = *(const v8bf*)(sbase + k0 + 32);
#endif
    }
    // A fragments (16x32 bf16): lane lo holds its row; K runs [hi*16,+15]
    v16bf a0, a1;
    {
      v8bf l0 = *(const v8bf*)(arow0 + k0 + hi * 16);
      v8bf h0 = *(const v8bf*)(arow0 + k0 + hi * 16 + 8);
      v8bf l1 = *(const v8bf*)(arow1 + k0 + hi * 16);
      v8bf h1 = *(const v8bf*)(arow1 + k0 + hi * 16 + 8);
      #pragma unroll
      for (int i = 0; i < 8; ++i) {
        a0[i] = l0[i]; a0[8 + i] = h0[i];
        a1[i] = l1[i]; a1[8 + i] = h1[i];
      }
    }
    #pragma unroll
    for (int g = 0; g < 4; ++g) {
      // B fragment (32x16): lane needs K = hi*16..+15 at fixed n (contiguous in LDS)
      v16bf bfrag = *(const v16bf*)&bt[buf][g * 16 + lo][hi * 16];
      acc[0][g] = WMMA_BF16(a0, bfrag, acc[0][g]);
      acc[1][g] = WMMA_BF16(a1, bfrag, acc[1][g]);
    }
#if HAVE_ASYNC_LDS
    __builtin_amdgcn_s_wait_asynccnt(0);
#endif
    __syncthreads();
  }

  #pragma unroll
  for (int rg = 0; rg < 2; ++rg)
    #pragma unroll
    for (int g = 0; g < 4; ++g) {
      const int n = colbase + g * 16 + lo;
      const float bv = bias[n];
      #pragma unroll
      for (int r = 0; r < 8; ++r)
        C[(size_t)(rowbase + rg * 16 + r + 8 * hi) * N + n] = acc[rg][g][r] + bv;
    }
}

// ---------------------------------------------------------------------------
// RoPE + qkv split: qkv f32 [B,S,H*192] -> q,k bf16 [B,H,S,64], vT bf16 [B,H,64,S]
// rot_dim=32, interleaved pairs; inv_freq_j = 10000^(-j/16)
// ---------------------------------------------------------------------------
__global__ void k_rope(const float* __restrict__ QKV, __bf16* __restrict__ QB,
                       __bf16* __restrict__ KB, __bf16* __restrict__ VT) {
  int idx = blockIdx.x * blockDim.x + threadIdx.x;
  if (idx >= Bn * Ss * Hh) return;
  const int h = idx % Hh;
  const int s = (idx / Hh) % Ss;
  const int b = idx / (Hh * Ss);
  const float* row = QKV + (size_t)(b * Ss + s) * N3 + h * (3 * HD);
  const size_t qo = ((size_t)(b * Hh + h) * Ss + s) * HD;
  __bf16* qd = QB + qo;
  __bf16* kd = KB + qo;
  #pragma unroll
  for (int j = 0; j < 16; ++j) {
    const float invf = expf(-(float)j * 0.5756462732485114f);  // ln(10000)/16
    float sn, cs;
    sincosf((float)s * invf, &sn, &cs);
    float q0 = row[2 * j], q1 = row[2 * j + 1];
    qd[2 * j]     = (__bf16)(q0 * cs - q1 * sn);
    qd[2 * j + 1] = (__bf16)(q1 * cs + q0 * sn);
    float ka = row[HD + 2 * j], kb = row[HD + 2 * j + 1];
    kd[2 * j]     = (__bf16)(ka * cs - kb * sn);
    kd[2 * j + 1] = (__bf16)(kb * cs + ka * sn);
  }
  for (int d = 32; d < HD; ++d) {
    qd[d] = (__bf16)row[d];
    kd[d] = (__bf16)row[HD + d];
  }
  __bf16* vd = VT + (size_t)(b * Hh + h) * HD * Ss;
  for (int d = 0; d < HD; ++d) vd[(size_t)d * Ss + s] = (__bf16)row[2 * HD + d];
}

// ---------------------------------------------------------------------------
// Attention with log_softmax: ao = (scale*QK^T + mask)@V - lse * colsum(V)
// Block = 128 threads = 4 waves; wave wv owns rows [bx*64 + 16*wv, +16) of (b,h)
// ---------------------------------------------------------------------------
__global__ void __launch_bounds__(128) k_attn(const __bf16* __restrict__ QB,
                                              const __bf16* __restrict__ KB,
                                              const __bf16* __restrict__ VT,
                                              const float* __restrict__ mask,
                                              __bf16* __restrict__ AOB) {
  __shared__ __bf16 sc[4][16][32];  // per-wave 16x32 score tile (row-major)
  const int lane = threadIdx.x & 31, wv = threadIdx.x >> 5;
  const int lo = lane & 15, hi = lane >> 4;
  const int bh = blockIdx.y;
  const int b = bh >> 4, h = bh & 15;
  const int rbase = blockIdx.x * 64 + wv * 16;
  const __bf16* q  = QB + (size_t)bh * Ss * HD;
  const __bf16* kk = KB + (size_t)bh * Ss * HD;
  const __bf16* vt = VT + (size_t)bh * HD * Ss;
  const float* mrow = mask + (size_t)b * Ss * Ss;

  // Q A-fragments for K=0..31 and K=32..63 (row m per lane, contiguous runs)
  const int m = rbase + lo;
  v16bf qa0, qa1;
  {
    v8bf t0 = *(const v8bf*)(q + (size_t)m * HD + hi * 16);
    v8bf t1 = *(const v8bf*)(q + (size_t)m * HD + 8 + hi * 16);
    v8bf t2 = *(const v8bf*)(q + (size_t)m * HD + 32 + hi * 16);
    v8bf t3 = *(const v8bf*)(q + (size_t)m * HD + 40 + hi * 16);
    #pragma unroll
    for (int i = 0; i < 8; ++i) {
      qa0[i] = t0[i]; qa0[8 + i] = t1[i];
      qa1[i] = t2[i]; qa1[8 + i] = t3[i];
    }
  }

  v8f acc[4];
  #pragma unroll
  for (int g = 0; g < 4; ++g)
    #pragma unroll
    for (int r = 0; r < 8; ++r) acc[g][r] = 0.f;
  float rmax[8], rsum[8], vsum[4];
  #pragma unroll
  for (int r = 0; r < 8; ++r) { rmax[r] = -3.0e38f; rsum[r] = 0.f; }
  #pragma unroll
  for (int g = 0; g < 4; ++g) vsum[g] = 0.f;

  for (int t0 = 0; t0 < Ss; t0 += 32) {
    // ---- scores for key cols [t0, t0+32): two 16-col WMMA tiles ----
    #pragma unroll
    for (int tc = 0; tc < 2; ++tc) {
      const int nt = t0 + tc * 16 + lo;  // key index (B-frag col)
      v16bf kb0 = *(const v16bf*)(kk + (size_t)nt * HD + hi * 16);
      v16bf kb1 = *(const v16bf*)(kk + (size_t)nt * HD + 32 + hi * 16);
      v8f s8;
      #pragma unroll
      for (int r = 0; r < 8; ++r) s8[r] = 0.f;
      s8 = WMMA_BF16(qa0, kb0, s8);
      s8 = WMMA_BF16(qa1, kb1, s8);
      float sv[8];
      #pragma unroll
      for (int r = 0; r < 8; ++r) {
        const int row = rbase + r + 8 * hi;
        float v = s8[r] * 0.125f + mrow[(size_t)row * Ss + nt];
        sv[r] = v;
        sc[wv][r + 8 * hi][tc * 16 + lo] = (__bf16)v;
      }
      // per-row (across 16 lanes of this half) online logsumexp update
      #pragma unroll
      for (int r = 0; r < 8; ++r) {
        float mx = sv[r];
        #pragma unroll
        for (int off = 8; off >= 1; off >>= 1) mx = fmaxf(mx, __shfl_xor(mx, off, 32));
        float e = expf(sv[r] - mx);
        #pragma unroll
        for (int off = 8; off >= 1; off >>= 1) e += __shfl_xor(e, off, 32);
        float nm = fmaxf(rmax[r], mx);
        rsum[r] = rsum[r] * expf(rmax[r] - nm) + e * expf(mx - nm);
        rmax[r] = nm;
      }
    }
    __syncthreads();
    // ---- acc += scores(16x32) @ V(32x64) ----
    v16bf sa;
    {
      v8bf s0 = *(const v8bf*)&sc[wv][lo][hi * 16];
      v8bf s1 = *(const v8bf*)&sc[wv][lo][hi * 16 + 8];
      #pragma unroll
      for (int i = 0; i < 8; ++i) { sa[i] = s0[i]; sa[8 + i] = s1[i]; }
    }
    #pragma unroll
    for (int g = 0; g < 4; ++g) {
      const int d = g * 16 + lo;
      v16bf vb = *(const v16bf*)(vt + (size_t)d * Ss + t0 + hi * 16);
      acc[g] = WMMA_BF16(sa, vb, acc[g]);
      float s = 0.f;
      #pragma unroll
      for (int i = 0; i < 16; ++i) s += (float)vb[i];
      vsum[g] += s;  // column-sum of V (this half's t range)
    }
    __syncthreads();
  }

  // combine halves of vsum, finalize lse, write ao = acc - lse * vsum
  #pragma unroll
  for (int g = 0; g < 4; ++g) vsum[g] += __shfl_xor(vsum[g], 16, 32);
  float lse[8];
  #pragma unroll
  for (int r = 0; r < 8; ++r) lse[r] = rmax[r] + logf(rsum[r]);
  #pragma unroll
  for (int g = 0; g < 4; ++g) {
    const int d = g * 16 + lo;
    #pragma unroll
    for (int r = 0; r < 8; ++r) {
      const int row = rbase + r + 8 * hi;
      AOB[(size_t)(b * Ss + row) * Dm + h * HD + d] = (__bf16)(acc[g][r] - lse[r] * vsum[g]);
    }
  }
}

// ---------------------------------------------------------------------------
// Host launch
// ---------------------------------------------------------------------------
extern "C" void kernel_launch(void* const* d_in, const int* in_sizes, int n_in,
                              void* d_out, int out_size, void* d_ws, size_t ws_size,
                              hipStream_t stream) {
  (void)in_sizes; (void)n_in; (void)out_size; (void)ws_size;
  const float* x    = (const float*)d_in[0];
  const float* mask = (const float*)d_in[1];
  const float* Win  = (const float*)d_in[2];
  const float* bin  = (const float*)d_in[3];
  const float* anw  = (const float*)d_in[4];
  const float* Wqkv = (const float*)d_in[5];
  const float* bqkv = (const float*)d_in[6];
  const float* Wout = (const float*)d_in[7];
  const float* bout = (const float*)d_in[8];
  const float* W1   = (const float*)d_in[9];
  const float* b1   = (const float*)d_in[10];
  const float* W2   = (const float*)d_in[11];
  const float* b2   = (const float*)d_in[12];
  const float* fnw  = (const float*)d_in[13];

  char* ws = (char*)d_ws;
  constexpr size_t MB = 1ull << 20;
  __bf16* WT_IN  = (__bf16*)(ws + 0 * MB);    // 2MB
  __bf16* WT_QKV = (__bf16*)(ws + 2 * MB);    // 6MB
  __bf16* WT_OUT = (__bf16*)(ws + 8 * MB);    // 2MB
  __bf16* WT_1   = (__bf16*)(ws + 10 * MB);   // 2MB
  __bf16* WT_2   = (__bf16*)(ws + 12 * MB);   // 2MB
  __bf16* XB     = (__bf16*)(ws + 14 * MB);   // 8MB
  float*  HPRE   = (float*)(ws + 22 * MB);    // 16MB (reused as H2 later)
  float*  Hbuf   = (float*)(ws + 38 * MB);    // 16MB
  __bf16* HB     = (__bf16*)(ws + 54 * MB);   // 8MB
  float*  QKVF   = (float*)(ws + 62 * MB);    // 48MB (reused as OF/F1/F2 later)
  __bf16* QBuf   = (__bf16*)(ws + 110 * MB);  // 8MB
  __bf16* KBuf   = (__bf16*)(ws + 118 * MB);  // 8MB
  __bf16* VTBuf  = (__bf16*)(ws + 126 * MB);  // 8MB
  __bf16* AOB    = (__bf16*)(ws + 134 * MB);  // 8MB
  __bf16* H2B    = (__bf16*)(ws + 142 * MB);  // 8MB
  __bf16* F1B    = (__bf16*)(ws + 150 * MB);  // 8MB  (total 158MB)
  // overlays (source buffers dead by the time these are written)
  float* OF = (float*)(ws + 62 * MB);   // over QKVF[0:16MB]
  float* F1 = (float*)(ws + 78 * MB);   // over QKVF[16:32MB]
  float* F2 = (float*)(ws + 94 * MB);   // over QKVF[32:48MB]
  float* H2 = (float*)(ws + 22 * MB);   // over HPRE

  const int nElem = Mrows * Dm;  // 4M
  const int EB = 256;
  const int gElem = (nElem + EB - 1) / EB;

  // weight transpose+convert, activation convert
  k_cvt<<<gElem, EB, 0, stream>>>(x, XB, nElem);
  k_wtrans<<<(Dm * Dm + EB - 1) / EB, EB, 0, stream>>>(Win, WT_IN, Dm, Dm);
  k_wtrans<<<(Dm * N3 + EB - 1) / EB, EB, 0, stream>>>(Wqkv, WT_QKV, Dm, N3);
  k_wtrans<<<(Dm * Dm + EB - 1) / EB, EB, 0, stream>>>(Wout, WT_OUT, Dm, Dm);
  k_wtrans<<<(Dm * Dm + EB - 1) / EB, EB, 0, stream>>>(W1, WT_1, Dm, Dm);
  k_wtrans<<<(Dm * Dm + EB - 1) / EB, EB, 0, stream>>>(W2, WT_2, Dm, Dm);

  // h = l2norm(x @ Win + bin) * attn_norm_w
  k_gemm<<<dim3(Dm / 64, Mrows / 256), 256, 0, stream>>>(XB, WT_IN, bin, HPRE, Mrows, Dm, Dm);
  k_l2norm<<<Mrows, 256, 0, stream>>>(HPRE, nullptr, anw, Hbuf, HB, Dm);

  // qkv = h @ Wqkv + bqkv ; RoPE + split
  k_gemm<<<dim3(N3 / 64, Mrows / 256), 256, 0, stream>>>(HB, WT_QKV, bqkv, QKVF, Mrows, N3, Dm);
  k_rope<<<(Bn * Ss * Hh + EB - 1) / EB, EB, 0, stream>>>(QKVF, QBuf, KBuf, VTBuf);

  // attention (log_softmax form)
  k_attn<<<dim3(Ss / 64, Bn * Hh), 128, 0, stream>>>(QBuf, KBuf, VTBuf, mask, AOB);

  // o = ao @ Wout + bout ; h2 = h + o
  k_gemm<<<dim3(Dm / 64, Mrows / 256), 256, 0, stream>>>(AOB, WT_OUT, bout, OF, Mrows, Dm, Dm);
  k_resid<<<gElem, EB, 0, stream>>>(Hbuf, OF, H2, H2B, nElem);

  // ffn = silu(h2 @ W1 + b1) @ W2 + b2
  k_gemm<<<dim3(Dm / 64, Mrows / 256), 256, 0, stream>>>(H2B, WT_1, b1, F1, Mrows, Dm, Dm);
  k_silu<<<gElem, EB, 0, stream>>>(F1, F1B, nElem);
  k_gemm<<<dim3(Dm / 64, Mrows / 256), 256, 0, stream>>>(F1B, WT_2, b2, F2, Mrows, Dm, Dm);

  // out = l2norm(h2 + ffn) * ffn_norm_w
  k_l2norm<<<Mrows, 256, 0, stream>>>(H2, F2, fnw, (float*)d_out, nullptr, Dm);
}
